// TTNetSharedParallelized_78013785964645
// MI455X (gfx1250) — matrix operand order repaired and checked by reading, not compiled, not run on hardware
//
#include <hip/hip_runtime.h>

// Problem dims (fixed by reference)
#define BGLOB 2048
#define NPIX  1024
#define MFEAT 4
#define TSTEP 64
#define RRANK 128
#define CCAT  10
#define BTILE 64
#define SSTR  136   // Slds row stride in bf16 elements (128 + 8 pad, keeps 16B alignment)

// LDS layout (dynamic):
//  Wpack : 4*8*4 fragments * 512 bf16  = 131072 B  (w_mid[c] in WMMA-B fragment-native order)
//  Slds  : 64 * 136 bf16               =  17408 B  (state tile, padded rows)
//  xs    : 256 f32                     =   1024 B  (x_t staging for this b-tile)
//  red   : 512 f32                     =   2048 B  (logit reduction scratch)
#define SMEM_WPACK 131072
#define SMEM_SLDS  17408
#define SMEM_XS    1024
#define SMEM_RED   2048
#define SMEM_BYTES (SMEM_WPACK + SMEM_SLDS + SMEM_XS + SMEM_RED)

typedef __attribute__((ext_vector_type(16))) __bf16 v16bf;
typedef __attribute__((ext_vector_type(8)))  __bf16 v8bf;
typedef __attribute__((ext_vector_type(8)))  float  v8f;
typedef __attribute__((ext_vector_type(4)))  float  v4f;

union BF16x16 { v16bf v; v8bf h[2]; };

// ---------------------------------------------------------------------------
// Kernel 1: trig feature map + 16-pixel averaging -> xdiv[t][b][m] (f32)
// ---------------------------------------------------------------------------
__global__ void ttnet_feat_kernel(const float* __restrict__ tensor,
                                  float* __restrict__ xdiv) {
  int gid = blockIdx.x * blockDim.x + threadIdx.x;
  if (gid >= BGLOB * TSTEP) return;
  int b = gid / TSTEP;
  int t = gid % TSTEP;
  const float* px = tensor + (size_t)b * NPIX + t * 16;
  float f0 = 0.f, f1 = 0.f, f2 = 0.f, f3 = 0.f;
#pragma unroll
  for (int p = 0; p < 16; ++p) {
    float x = px[p];
    float a = 1.5707963267948966f * x;
    float c = cosf(a), s = sinf(a);
    float cc = c * c, ss = s * s;
    f0 += cc * c;   // cos^3
    f1 += cc * s;   // cos^2 sin
    f2 += c * ss;   // cos sin^2
    f3 += ss * s;   // sin^3
  }
  float inv = 1.0f / 16.0f;
  float* o = xdiv + (size_t)t * (BGLOB * MFEAT) + (size_t)b * MFEAT;
  o[0] = f0 * inv; o[1] = f1 * inv; o[2] = f2 * inv; o[3] = f3 * inv;
}

// ---------------------------------------------------------------------------
// Kernel 2: persistent per-(c, b-tile) TT scan, 64 steps, all in LDS/registers.
// 512 threads = 16 waves = 4 row-tiles (rt) x 4 column-groups (cg, 32 cols).
// Each step: Y[m] = S @ Wmid[c,:,m,:]  via v_wmma_f32_16x16x32_bf16,
// then state'[b,o] = sum_m x[t][b][m] * Y[m][b,o]  (register FMAs), bf16 -> LDS.
// ---------------------------------------------------------------------------
__global__ __launch_bounds__(512)
void ttnet_scan_kernel(const float* __restrict__ wfirst,
                       const float* __restrict__ wmid,
                       const float* __restrict__ wlast,
                       const float* __restrict__ xdiv,
                       float* __restrict__ logits) {
  extern __shared__ char smem[];
  __bf16* Wpack = (__bf16*)smem;
  __bf16* Slds  = (__bf16*)(smem + SMEM_WPACK);
  float*  xs    = (float*)(smem + SMEM_WPACK + SMEM_SLDS);
  float*  red   = (float*)(smem + SMEM_WPACK + SMEM_SLDS + SMEM_XS);

  const int tid  = threadIdx.x;
  const int c    = blockIdx.y;
  const int b0   = blockIdx.x * BTILE;
  const int wave = tid >> 5;
  const int lane = tid & 31;
  const int rt   = wave & 3;   // row tile: rows rt*16 .. rt*16+15
  const int cg   = wave >> 2;  // column group: cols cg*32 .. cg*32+31 of each Y[m]
  const int lh   = lane >> 4;  // lane half
  const int l16  = lane & 15;

  // ---- pack w_mid[c] (f32, layout [i][m][o]) into WMMA-B fragments (bf16) ----
  // Fragment (m, nt, kt): 32x16 (KxN), K = i. Lane L holds col o = nt*16+L%16,
  // K = kt*32 + (L/16)*16 + e, e=0..15.  Stored contiguously: frag*512 + L*16 + e.
  {
    const float* wm = wmid + (size_t)c * (RRANK * MFEAT * RRANK);
    for (int q = tid; q < RRANK * MFEAT * RRANK; q += 512) {
      int o = q & 127;
      int r = q >> 7;
      int m = r & 3;
      int k = r >> 2;                        // i index (K dim of GEMM)
      float v = wm[(size_t)k * 512 + m * 128 + o];
      int kt = k >> 5, kk = k & 31;
      int nt = o >> 4;
      int L  = (o & 15) | (((kk >> 4) & 1) << 4);
      int e  = kk & 15;
      int fragId = (m * 8 + nt) * 4 + kt;
      Wpack[fragId * 512 + L * 16 + e] = (__bf16)v;
    }
  }
  // ---- initial state: state[b,o] = sum_m w_first[c,0,m,o]  (b-independent) ----
  {
    int o = tid & 127, rb = tid >> 7;
    const float* wf = wfirst + (size_t)c * (MFEAT * RRANK);
    float s = wf[o] + wf[128 + o] + wf[256 + o] + wf[384 + o];
    __bf16 sb = (__bf16)s;
    for (int rr = 0; rr < 16; ++rr) Slds[(rb * 16 + rr) * SSTR + o] = sb;
  }
  __syncthreads();

  const int arow = rt * 16 + l16;  // A-fragment row for this lane

  for (int t = 0; t < TSTEP; ++t) {
    // stage x_t tile (64 rows x 4) into LDS
    if (tid < 256) xs[tid] = xdiv[(size_t)t * (BGLOB * MFEAT) + b0 * MFEAT + tid];

    // load A fragments (state rows, shared across all m and cg) from Slds
    BF16x16 A[4];
#pragma unroll
    for (int kt = 0; kt < 4; ++kt) {
      const __bf16* ap = Slds + arow * SSTR + kt * 32 + (lh ? 16 : 0);
      A[kt].h[0] = *(const v8bf*)ap;          // K = base .. base+7
      A[kt].h[1] = *(const v8bf*)(ap + 8);    // K = base+8 .. base+15
    }
    __syncthreads();  // A reads + xs writes done before state overwrite / xs reads

    // GEMM: acc[m][j] = 16x16 tile of S @ Wmid[:, m, :], cols (cg*2+j)*16..
    v8f acc[4][2];
#pragma unroll
    for (int m = 0; m < 4; ++m) {
#pragma unroll
      for (int j = 0; j < 2; ++j) {
        v8f a = {};
#pragma unroll
        for (int kt = 0; kt < 4; ++kt) {
          int fragId = (m * 8 + (cg * 2 + j)) * 4 + kt;
          BF16x16 Bf;
          const v8bf* bp = (const v8bf*)(Wpack + fragId * 512 + lane * 16);
          Bf.h[0] = bp[0];
          Bf.h[1] = bp[1];
          a = __builtin_amdgcn_wmma_f32_16x16x32_bf16(
              false, A[kt].v, false, Bf.v, (short)0, a, false, false);
        }
        acc[m][j] = a;
      }
    }

    // combine over m with x_t and write new bf16 state
#pragma unroll
    for (int v = 0; v < 8; ++v) {
      int row = rt * 16 + v + 8 * lh;            // C/D layout: vgpr v, lane half
      v4f xv = *(const v4f*)(xs + row * 4);      // x[t][b0+row][0..3]
#pragma unroll
      for (int j = 0; j < 2; ++j) {
        float val = xv[0] * acc[0][j][v] + xv[1] * acc[1][j][v] +
                    xv[2] * acc[2][j][v] + xv[3] * acc[3][j][v];
        Slds[row * SSTR + cg * 32 + j * 16 + l16] = (__bf16)val;
      }
    }
    __syncthreads();  // state writes done before next step's A reads
  }

  // ---- final: logit[c][b] = sum_i state[b,i] * (sum_m w_last[c,i,m,0]) ----
  {
    int rrow = tid >> 3, seg = tid & 7;
    const float* wl = wlast + (size_t)c * (RRANK * MFEAT);
    float partial = 0.f;
#pragma unroll 4
    for (int ii = 0; ii < 16; ++ii) {
      int i = seg * 16 + ii;
      float w = wl[i * 4 + 0] + wl[i * 4 + 1] + wl[i * 4 + 2] + wl[i * 4 + 3];
      partial += (float)Slds[rrow * SSTR + i] * w;
    }
    red[tid] = partial;
  }
  __syncthreads();
  if (tid < 64) {
    float s = 0.f;
#pragma unroll
    for (int k = 0; k < 8; ++k) s += red[tid * 8 + k];
    logits[(size_t)c * BGLOB + b0 + tid] = s;
  }
}

// ---------------------------------------------------------------------------
// Kernel 3: log_softmax over C=10 per batch row, transpose to (B, C)
// ---------------------------------------------------------------------------
__global__ void ttnet_lsm_kernel(const float* __restrict__ logits,
                                 float* __restrict__ out) {
  int b = blockIdx.x * blockDim.x + threadIdx.x;
  if (b >= BGLOB) return;
  float L[CCAT];
  float mx = -3.0e38f;
#pragma unroll
  for (int c = 0; c < CCAT; ++c) {
    L[c] = logits[(size_t)c * BGLOB + b];
    mx = fmaxf(mx, L[c]);
  }
  float s = 0.f;
#pragma unroll
  for (int c = 0; c < CCAT; ++c) s += expf(L[c] - mx);
  float lse = mx + logf(s);
#pragma unroll
  for (int c = 0; c < CCAT; ++c) out[(size_t)b * CCAT + c] = L[c] - lse;
}

// ---------------------------------------------------------------------------
extern "C" void kernel_launch(void* const* d_in, const int* in_sizes, int n_in,
                              void* d_out, int out_size, void* d_ws, size_t ws_size,
                              hipStream_t stream) {
  const float* tensor = (const float*)d_in[0];  // (B, N)
  const float* wfirst = (const float*)d_in[1];  // (C, 1, M, R)
  const float* wmid   = (const float*)d_in[2];  // (C, R, M, R)
  const float* wlast  = (const float*)d_in[3];  // (C, R, M, 1)
  float* out = (float*)d_out;                   // (B, C)

  float* xdiv   = (float*)d_ws;                                   // T*B*M f32 = 2 MB
  float* logits = (float*)((char*)d_ws +
                           (size_t)TSTEP * BGLOB * MFEAT * sizeof(float)); // C*B f32

  (void)hipFuncSetAttribute((const void*)ttnet_scan_kernel,
                            hipFuncAttributeMaxDynamicSharedMemorySize, SMEM_BYTES);

  int nfeat = BGLOB * TSTEP;
  ttnet_feat_kernel<<<(nfeat + 255) / 256, 256, 0, stream>>>(tensor, xdiv);

  dim3 grid(BGLOB / BTILE, CCAT);  // 32 x 10 = 320 workgroups
  ttnet_scan_kernel<<<grid, 512, SMEM_BYTES, stream>>>(wfirst, wmid, wlast, xdiv, logits);

  ttnet_lsm_kernel<<<(BGLOB + 255) / 256, 256, 0, stream>>>(logits, out);
}
